// SoftAttentionMax_51281909514278
// MI455X (gfx1250) — matrix operand order repaired and checked by reading, not compiled, not run on hardware
//
#include <hip/hip_runtime.h>
#include <hip/hip_bf16.h>

// ---------------------------------------------------------------------------
// SAGAN self-attention for MI455X (gfx1250, wave32, WMMA).
// B=16, H=W=64 (hw=4096, pooled 1024), C=512, Cfg=64, Ch=256.
// All matmuls via v_wmma_f32_16x16x32_f16; softmax fused flash-style so the
// [16,4096,1024] score tensor is never materialized. Packed-weight GEMMs are
// templated on tile counts so B-fragment loads fold into immediate offsets.
// ---------------------------------------------------------------------------

typedef __attribute__((ext_vector_type(16))) _Float16 v16h;
typedef __attribute__((ext_vector_type(8)))  float    v8f;

#define WMMA_F16(a, b, c) \
  __builtin_amdgcn_wmma_f32_16x16x32_f16(false, (a), false, (b), (short)0, (c), false, false)

static constexpr int kB    = 16;
static constexpr int kHW   = 4096;   // 64*64
static constexpr int kC    = 512;
static constexpr int kCfg  = 64;
static constexpr int kCh   = 256;

// -------- fragment load helpers (16B aligned halves) ------------------------
__device__ __forceinline__ v16h ld_frag(const _Float16* lo, const _Float16* hi) {
  uint4 a = *(const uint4*)lo;
  uint4 b = *(const uint4*)hi;
  v16h r;
  ((uint4*)&r)[0] = a;
  ((uint4*)&r)[1] = b;
  return r;
}
__device__ __forceinline__ v16h ld_frag32(const _Float16* p) { return ld_frag(p, p + 8); }

// ---------------------------------------------------------------------------
// Templated inner GEMM over a 16-row LDS strip (A) and packed weights (B).
// NT = N/16 tile count (compile-time -> B loads get immediate offsets),
// KCH = K/32 chunk count. pwBase = pw + nt*512 + lane*16.
// ---------------------------------------------------------------------------
template <int NT, int KCH>
__device__ __forceinline__ v8f gemm_strip(const _Float16* __restrict__ rowp,
                                          const _Float16* __restrict__ pwBase,
                                          int koff) {
  v8f acc = {};
#pragma unroll
  for (int kci = 0; kci < KCH; ++kci) {
    v16h a = ld_frag(rowp + kci * 32 + koff, rowp + kci * 32 + 16 + koff);
    v16h b = ld_frag32(pwBase + (size_t)kci * NT * 512);
    acc = WMMA_F16(a, b, acc);
  }
  return acc;
}

// ---------------------------------------------------------------------------
// Pack a row-major f32 weight [K, N] into WMMA B-fragment order (f16).
// Fragment (kt, nt): lane holds column n = nt*16 + lane%16,
// halves h: K = kt*32 + (lane/16)*8 + h + (h&8). 32 B contiguous per lane.
// ---------------------------------------------------------------------------
__global__ void pack_w_kernel(const float* __restrict__ W, _Float16* __restrict__ dst,
                              int K, int N) {
  int t = blockIdx.x * blockDim.x + threadIdx.x;
  int nTiles = N >> 4;
  int nFrags = (K >> 5) * nTiles;
  if (t >= nFrags * 32) return;
  int lane = t & 31, frag = t >> 5;
  int kt = frag / nTiles, nt = frag % nTiles;
  int n = nt * 16 + (lane & 15);
  int kbase = kt * 32 + (lane >> 4) * 8;
  _Float16 tmp[16];
#pragma unroll
  for (int h = 0; h < 16; ++h) tmp[h] = (_Float16)W[(size_t)(kbase + h + (h & 8)) * N + n];
  _Float16* out = dst + (size_t)frag * 512 + lane * 16;
#pragma unroll
  for (int h = 0; h < 16; ++h) out[h] = tmp[h];
}

// ---------------------------------------------------------------------------
// Fused f/g/h projection: one block stages 32 pixels x 512 channels of x into
// LDS (f16) once; 8 waves compute 48 output tiles (g:4, f:4, h:16 N-tiles,
// each x 2 M-tiles), 16 WMMA K-steps per tile. Outputs row-major f16.
// ---------------------------------------------------------------------------
__global__ __launch_bounds__(256) void proj_kernel(
    const float* __restrict__ x,
    const _Float16* __restrict__ pWf, const _Float16* __restrict__ pWg,
    const _Float16* __restrict__ pWh,
    const float* __restrict__ bf, const float* __restrict__ bg,
    const float* __restrict__ bh,
    _Float16* __restrict__ f_full, _Float16* __restrict__ g_out,
    _Float16* __restrict__ h_full) {
  __shared__ __align__(16) _Float16 xs[32][520];  // 520*2 B rows keep 16B align
  const int p0 = blockIdx.x * 32;                 // pixel row base in [0, B*hw)
  const float* xg = x + (size_t)p0 * kC;
  for (int i = threadIdx.x; i < 32 * kC; i += 256) {
    xs[i >> 9][i & 511] = (_Float16)xg[i];
  }
  __syncthreads();

  const int wave = threadIdx.x >> 5, lane = threadIdx.x & 31;
  const int koff = (lane >> 4) * 8;
  const int lane16 = lane & 15;

  for (int job = wave; job < 48; job += 8) {
    const int mt = job & 1, t = job >> 1;
    const _Float16* rowp = xs[mt * 16 + lane16];
    v8f acc;
    const float* bias;
    _Float16* dst;
    int nt, N;
    if (t < 4) {            // g projection (N=64, 4 tiles, K=512)
      nt = t; N = kCfg; bias = bg; dst = g_out;
      acc = gemm_strip<4, 16>(rowp, pWg + (size_t)nt * 512 + lane * 16, koff);
    } else if (t < 8) {     // f projection
      nt = t - 4; N = kCfg; bias = bf; dst = f_full;
      acc = gemm_strip<4, 16>(rowp, pWf + (size_t)nt * 512 + lane * 16, koff);
    } else {                // h projection (N=256, 16 tiles, K=512)
      nt = t - 8; N = kCh; bias = bh; dst = h_full;
      acc = gemm_strip<16, 16>(rowp, pWh + (size_t)nt * 512 + lane * 16, koff);
    }
    const int colg = nt * 16 + lane16;
    const float bv = bias[colg];
    const int rbase = p0 + mt * 16 + (lane >> 4) * 8;
#pragma unroll
    for (int r = 0; r < 8; ++r)
      dst[(size_t)(rbase + r) * N + colg] = (_Float16)(acc[r] + bv);
  }
}

// ---------------------------------------------------------------------------
// 2x2 maxpool of f_full -> fragment-packed keys for s = g @ f_p^T.
// Fragment (b, jt, kci): lane n = key = jt*16+lane%16; halves = d dims.
// ---------------------------------------------------------------------------
__global__ void fp_pack_kernel(const _Float16* __restrict__ f_full,
                               _Float16* __restrict__ fpPack) {
  int t = blockIdx.x * blockDim.x + threadIdx.x;  // 16*64*2*32 threads
  int lane = t & 31, frag = t >> 5;
  int kci = frag & 1, jt = (frag >> 1) & 63, b = frag >> 7;
  int key = jt * 16 + (lane & 15);
  int qh = key >> 5, qw = key & 31;
  int n00 = qh * 128 + qw * 2;
  const _Float16* base = f_full + (size_t)(b * kHW) * kCfg;
  const _Float16* r0 = base + (size_t)n00 * kCfg;
  const _Float16* r1 = base + (size_t)(n00 + 1) * kCfg;
  const _Float16* r2 = base + (size_t)(n00 + 64) * kCfg;
  const _Float16* r3 = base + (size_t)(n00 + 65) * kCfg;
  int d0 = kci * 32 + (lane >> 4) * 8;
  _Float16 tmp[16];
#pragma unroll
  for (int h = 0; h < 16; ++h) {
    int d = d0 + h + (h & 8);
    float m = fmaxf(fmaxf((float)r0[d], (float)r1[d]),
                    fmaxf((float)r2[d], (float)r3[d]));
    tmp[h] = (_Float16)m;
  }
  _Float16* out = fpPack + (size_t)frag * 512 + lane * 16;
#pragma unroll
  for (int h = 0; h < 16; ++h) out[h] = tmp[h];
}

// ---------------------------------------------------------------------------
// 2x2 maxpool of h_full -> fragment-packed values for o = beta @ h_p.
// Fragment (b, kc, nt): lane n = channel = nt*16+lane%16; halves = keys.
// ---------------------------------------------------------------------------
__global__ void hp_pack_kernel(const _Float16* __restrict__ h_full,
                               _Float16* __restrict__ hpPack) {
  int t = blockIdx.x * blockDim.x + threadIdx.x;  // 16*32*16*32 threads
  int lane = t & 31, frag = t >> 5;
  int nt = frag & 15, kc = (frag >> 4) & 31, b = frag >> 9;
  int n = nt * 16 + (lane & 15);
  int key0 = kc * 32 + (lane >> 4) * 8;
  const _Float16* base = h_full + (size_t)(b * kHW) * kCh;
  _Float16 tmp[16];
#pragma unroll
  for (int h = 0; h < 16; ++h) {
    int key = key0 + h + (h & 8);
    int qh = key >> 5, qw = key & 31;
    int n00 = qh * 128 + qw * 2;
    float m = fmaxf(
        fmaxf((float)base[(size_t)n00 * kCh + n], (float)base[(size_t)(n00 + 1) * kCh + n]),
        fmaxf((float)base[(size_t)(n00 + 64) * kCh + n], (float)base[(size_t)(n00 + 65) * kCh + n]));
    tmp[h] = (_Float16)m;
  }
  _Float16* out = hpPack + (size_t)frag * 512 + lane * 16;
#pragma unroll
  for (int h = 0; h < 16; ++h) out[h] = tmp[h];
}

// ---------------------------------------------------------------------------
// Flash attention: each wave owns a 16-query tile, streams 1024 keys in
// 32-key chunks. s-tiles via WMMA -> LDS -> online softmax (row lanes) ->
// reload p as A fragment -> 16 value WMMAs into the 16x256 accumulator.
// ---------------------------------------------------------------------------
__global__ __launch_bounds__(256) void attn_kernel(
    const _Float16* __restrict__ g, const _Float16* __restrict__ fpPack,
    const _Float16* __restrict__ hpPack, _Float16* __restrict__ o_out) {
  __shared__ __align__(16) float    sS[8][16][33];
  __shared__ __align__(16) _Float16 sP[8][16][32];
  __shared__ float sStat[8][16];

  const int b = blockIdx.x >> 5;
  const int qblk = blockIdx.x & 31;
  const int wave = threadIdx.x >> 5, lane = threadIdx.x & 31;
  const int q0 = qblk * 128 + wave * 16;
  const size_t grow = (size_t)b * kHW + q0;
  const int koff = (lane >> 4) * 8;
  const int rb = (lane >> 4) * 8;

  // Query A fragments (K=64 -> two K-chunks), kept in registers all loop.
  const _Float16* gp = g + (grow + (lane & 15)) * kCfg;
  const v16h a0 = ld_frag(gp + koff, gp + 16 + koff);
  const v16h a1 = ld_frag(gp + 32 + koff, gp + 48 + koff);

  const _Float16* fpB = fpPack + (size_t)b * 128 * 512 + lane * 16;
  const _Float16* hpB = hpPack + (size_t)b * 32 * 16 * 512 + lane * 16;

  v8f oacc[16];
#pragma unroll
  for (int nt = 0; nt < 16; ++nt) oacc[nt] = 0.f;
  float m_r = -3.0e38f, l_r = 0.f;  // row stats owned by lanes 0..15

  for (int kc = 0; kc < 32; ++kc) {
    // ---- scores for 32 keys (two 16-key tiles) ----
    const _Float16* fpc = fpB + (size_t)kc * 4 * 512;
#pragma unroll
    for (int t2 = 0; t2 < 2; ++t2) {
      v16h b0 = ld_frag32(fpc + (size_t)(t2 * 2 + 0) * 512);
      v16h b1 = ld_frag32(fpc + (size_t)(t2 * 2 + 1) * 512);
      v8f s = {};
      s = WMMA_F16(a0, b0, s);
      s = WMMA_F16(a1, b1, s);
#pragma unroll
      for (int r = 0; r < 8; ++r)
        sS[wave][rb + r][t2 * 16 + (lane & 15)] = s[r];
    }
    __syncthreads();

    // ---- online softmax: lane r owns query row r ----
    if (lane < 16) {
      float mx = m_r;
#pragma unroll
      for (int k = 0; k < 32; ++k) mx = fmaxf(mx, sS[wave][lane][k]);
      float sc = __expf(m_r - mx);
      float ps = 0.f;
#pragma unroll
      for (int k = 0; k < 32; ++k) {
        float p = __expf(sS[wave][lane][k] - mx);
        sP[wave][lane][k] = (_Float16)p;
        ps += p;
      }
      l_r = l_r * sc + ps;
      m_r = mx;
      sStat[wave][lane] = sc;
    }
    __syncthreads();

    // ---- rescale accumulators, then p @ h_p ----
    float scv[8];
#pragma unroll
    for (int r = 0; r < 8; ++r) scv[r] = sStat[wave][rb + r];
#pragma unroll
    for (int nt = 0; nt < 16; ++nt)
#pragma unroll
      for (int r = 0; r < 8; ++r) oacc[nt][r] *= scv[r];

    const _Float16* pr = sP[wave][lane & 15];
    v16h pa = ld_frag(pr + koff, pr + 16 + koff);
    const _Float16* hpc = hpB + (size_t)kc * 16 * 512;
    __builtin_prefetch(hpc + 16 * 512, 0, 0);
#pragma unroll
    for (int nt = 0; nt < 16; ++nt) {
      v16h bv = ld_frag32(hpc + (size_t)nt * 512);
      oacc[nt] = WMMA_F16(pa, bv, oacc[nt]);
    }
    __syncthreads();
  }

  if (lane < 16) sStat[wave][lane] = l_r;
  __syncthreads();
  float inv[8];
#pragma unroll
  for (int r = 0; r < 8; ++r) inv[r] = 1.0f / sStat[wave][rb + r];
#pragma unroll
  for (int nt = 0; nt < 16; ++nt) {
    int colg = nt * 16 + (lane & 15);
#pragma unroll
    for (int r = 0; r < 8; ++r)
      o_out[(grow + rb + r) * kCh + colg] = (_Float16)(oacc[nt][r] * inv[r]);
  }
}

// ---------------------------------------------------------------------------
// Final projection + residual: out = gamma * (o @ Wo + bo) + x  (f32 out).
// ---------------------------------------------------------------------------
__global__ __launch_bounds__(256) void outproj_kernel(
    const _Float16* __restrict__ o, const _Float16* __restrict__ pWo,
    const float* __restrict__ bo, const float* __restrict__ x,
    const float* __restrict__ gamma, float* __restrict__ out) {
  __shared__ __align__(16) _Float16 os[32][264];  // 264*2 B rows, 16B aligned
  const int p0 = blockIdx.x * 32;
  const _Float16* og = o + (size_t)p0 * kCh;
  for (int i = threadIdx.x; i < 32 * kCh; i += 256)
    os[i >> 8][i & 255] = og[i];
  __syncthreads();

  const int wave = threadIdx.x >> 5, lane = threadIdx.x & 31;
  const int koff = (lane >> 4) * 8;
  const int lane16 = lane & 15;
  const float gm = gamma[0];

  for (int job = wave; job < 64; job += 8) {
    const int mt = job & 1, nt = job >> 1;  // nt in [0,32): 512 output channels
    const _Float16* rowp = os[mt * 16 + lane16];
    // K=256 -> 8 chunks; 32 N-tiles (compile-time) -> immediate offsets
    v8f acc = gemm_strip<32, 8>(rowp, pWo + (size_t)nt * 512 + lane * 16, koff);
    const int colg = nt * 16 + lane16;
    const float bv = bo[colg];
    const int rbase = p0 + mt * 16 + (lane >> 4) * 8;
#pragma unroll
    for (int r = 0; r < 8; ++r) {
      size_t idx = (size_t)(rbase + r) * kC + colg;
      out[idx] = gm * (acc[r] + bv) + x[idx];
    }
  }
}

// ---------------------------------------------------------------------------
extern "C" void kernel_launch(void* const* d_in, const int* in_sizes, int n_in,
                              void* d_out, int out_size, void* d_ws, size_t ws_size,
                              hipStream_t stream) {
  const float* x     = (const float*)d_in[0];
  const float* Wf    = (const float*)d_in[1];
  const float* bf    = (const float*)d_in[2];
  const float* Wg    = (const float*)d_in[3];
  const float* bg    = (const float*)d_in[4];
  const float* Wh    = (const float*)d_in[5];
  const float* bh    = (const float*)d_in[6];
  const float* Wo    = (const float*)d_in[7];
  const float* bo    = (const float*)d_in[8];
  const float* gamma = (const float*)d_in[9];
  float* out = (float*)d_out;

  // ---- carve workspace ----
  char* p = (char*)d_ws;
  auto carve = [&](size_t bytes) {
    char* r = p;
    p += (bytes + 255) & ~size_t(255);
    return r;
  };
  _Float16* pWf    = (_Float16*)carve(64  * 1024);                     // 64 frags
  _Float16* pWg    = (_Float16*)carve(64  * 1024);
  _Float16* pWh    = (_Float16*)carve(256 * 1024);
  _Float16* pWo    = (_Float16*)carve(256 * 1024);
  _Float16* g_buf  = (_Float16*)carve((size_t)kB * kHW * kCfg * 2);
  _Float16* f_full = (_Float16*)carve((size_t)kB * kHW * kCfg * 2);
  _Float16* h_full = (_Float16*)carve((size_t)kB * kHW * kCh  * 2);
  _Float16* fpPack = (_Float16*)carve((size_t)kB * 128 * 1024);        // 2 MB
  _Float16* hpPack = (_Float16*)carve((size_t)kB * 512 * 1024);        // 8 MB
  _Float16* o_buf  = (_Float16*)carve((size_t)kB * kHW * kCh  * 2);

  // ---- 1. pack weights into B-fragment order ----
  pack_w_kernel<<<(64  * 32 + 255) / 256, 256, 0, stream>>>(Wf, pWf, kC,  kCfg);
  pack_w_kernel<<<(64  * 32 + 255) / 256, 256, 0, stream>>>(Wg, pWg, kC,  kCfg);
  pack_w_kernel<<<(256 * 32 + 255) / 256, 256, 0, stream>>>(Wh, pWh, kC,  kCh);
  pack_w_kernel<<<(256 * 32 + 255) / 256, 256, 0, stream>>>(Wo, pWo, kCh, kC);

  // ---- 2. fused f/g/h projections ----
  proj_kernel<<<(kB * kHW) / 32, 256, 0, stream>>>(
      x, pWf, pWg, pWh, bf, bg, bh, f_full, g_buf, h_full);

  // ---- 3. maxpool + fragment pack of keys/values ----
  fp_pack_kernel<<<(kB * 128 * 32) / 256, 256, 0, stream>>>(f_full, fpPack);
  hp_pack_kernel<<<(kB * 512 * 32) / 256, 256, 0, stream>>>(h_full, hpPack);

  // ---- 4. flash attention ----
  attn_kernel<<<kB * 32, 256, 0, stream>>>(g_buf, fpPack, hpPack, o_buf);

  // ---- 5. output projection + residual ----
  outproj_kernel<<<(kB * kHW) / 32, 256, 0, stream>>>(o_buf, pWo, bo, x, gamma, out);
}
